// QRNNLayer_26611617366442
// MI455X (gfx1250) — matrix operand order repaired and checked by reading, not compiled, not run on hardware
//
#include <hip/hip_runtime.h>
#include <hip/hip_bf16.h>

typedef __bf16 bf16_t;
typedef __attribute__((ext_vector_type(16))) __bf16 v16bf;
typedef __attribute__((ext_vector_type(8)))  float  v8f;
typedef __attribute__((ext_vector_type(4)))  __bf16 v4bf;

static constexpr int SEQ_   = 1024;
static constexpr int BATCH_ = 16;
static constexpr int IN_    = 1024;
static constexpr int HID_   = 1024;
static constexpr int KDIM_  = 2048;                 // IN * (LOOKBACK+1)
static constexpr int LANES_ = BATCH_ * HID_;        // 16384 recurrence lanes
static constexpr int NCHUNK_ = 64;                  // scan chunks
static constexpr int LCH_    = SEQ_ / NCHUNK_;      // 16 steps per chunk

// interleaved LDS layout for the A panel (see qrnn_gemm)
static constexpr int RSTRIDE = 80;                  // bytes per row: 64 data + 16 pad (bank rotate)
static constexpr int SSTRIDE = 16 * RSTRIDE;        // 1280 bytes per slice within a ko-group
static constexpr int GSTRIDE = 5 * SSTRIDE + 64;    // 6464 bytes per ko-group (pad rotates banks)
static constexpr int LDS_BYTES = 32 * GSTRIDE;      // 206848 (< 320 KB per WGP)

// workspace layout (bytes)
static constexpr size_t XB_OFF  = 0;                 // bf16 [SEQ*BATCH*IN]   32 MB
static constexpr size_t WPZ_OFF = 33554432;          // bf16 [HID*KDIM]        4 MB
static constexpr size_t WPF_OFF = 37748736;          // bf16 [HID*KDIM]        4 MB
static constexpr size_t ZA_OFF  = 41943040;          // bf16 [SEQ*LANES]      32 MB
static constexpr size_t FA_OFF  = 75497472;          // bf16 [SEQ*LANES]      32 MB
static constexpr size_t AC_OFF  = 109051904;         // f32  [NCHUNK*LANES]    4 MB
static constexpr size_t BC_OFF  = 113246208;         // f32  [NCHUNK*LANES]    4 MB
static constexpr size_t SS_OFF  = 117440512;         // f32  [NCHUNK*LANES]    4 MB

// ---------------------------------------------------------------- pack X -> bf16
__global__ __launch_bounds__(256) void qrnn_pack_x(const float* __restrict__ X,
                                                   bf16_t* __restrict__ Xb) {
    const int g = blockIdx.x * 256 + threadIdx.x;          // 4 elems per thread
    const float4 v = ((const float4*)X)[g];
    v4bf o;
    o[0] = (__bf16)v.x; o[1] = (__bf16)v.y; o[2] = (__bf16)v.z; o[3] = (__bf16)v.w;
    *(v4bf*)(Xb + (size_t)g * 4) = o;
}

// ---------------------------------------------------------------- pack W (H,IN,2) -> bf16 Wp[h][kk], kk = k*1024 + c
__global__ __launch_bounds__(256) void qrnn_pack_w(const float* __restrict__ Wz,
                                                   const float* __restrict__ Wf,
                                                   bf16_t* __restrict__ Wpz,
                                                   bf16_t* __restrict__ Wpf) {
    const int g = blockIdx.x * 256 + threadIdx.x;          // 0..2097151
    const float*  src = blockIdx.y ? Wf  : Wz;
    bf16_t*       dst = blockIdx.y ? Wpf : Wpz;
    const int h  = g >> 11;
    const int kk = g & 2047;
    const int c  = kk & 1023;
    const int k  = kk >> 10;
    dst[(size_t)h * KDIM_ + kk] = (__bf16)src[(size_t)h * KDIM_ + c * 2 + k];
}

// ---------------------------------------------------------------- fused Z/F GEMM + bias + activations
// grid (8, 256): block = 4 timesteps (64 rows) x 128 cols; 8 waves, each wave:
// 16 cols x 4 M-tiles, B fragments (z & f) held in regs across the 4 tiles.
//
// LDS A panel is k-interleaved: [ko_group g=K/32][slice s=0..4][row 0..15][64B of K=32g..32g+31]
// so all 8 fragment loads per k-step are immediate offsets off ONE base bumped by GSTRIDE.
__global__ __launch_bounds__(256) void qrnn_gemm(const bf16_t* __restrict__ Xb,
                                                 const bf16_t* __restrict__ Wpz,
                                                 const bf16_t* __restrict__ Wpf,
                                                 const float*  __restrict__ bz,
                                                 const float*  __restrict__ bfv,
                                                 bf16_t* __restrict__ Zact,
                                                 bf16_t* __restrict__ Fact) {
    extern __shared__ char smem[];
    const int tid = threadIdx.x;
    const int t0  = blockIdx.y * 4;

    // stage X[t0-1 .. t0+3]: slice s holds time t0-1+s (16 rows x 1024 bf16), scattered
    // into the k-interleaved layout.  10240 16-byte chunks, 40 per thread.
    for (int it = 0; it < 40; ++it) {
        const int linear = tid + it * 256;                  // chunk id
        const int slice  = linear >> 11;                    // 2048 chunks per slice
        const int rem    = linear & 2047;
        const int row    = rem >> 7;                        // 128 chunks per row
        const int cidx   = rem & 127;                       // 16B chunk within row (K = 8*cidx)
        const int ts     = t0 - 1 + slice;
        uint4 v = {0u, 0u, 0u, 0u};
        if (ts >= 0)
            v = *(const uint4*)((const char*)Xb + (size_t)ts * (BATCH_ * IN_ * 2)
                                + row * 2048 + cidx * 16);
        const int loff = (cidx >> 2) * GSTRIDE + slice * SSTRIDE + row * RSTRIDE + (cidx & 3) * 16;
        *(uint4*)(smem + loff) = v;
    }
    __syncthreads();

    const int wave = tid >> 5;
    const int lane = tid & 31;
    const int l    = lane & 15;
    const int lh   = lane >> 4;
    const int col  = blockIdx.x * 128 + wave * 16 + l;

    // B (32x16 bf16): lane covers K = 8*lh+[0..7] and 16+8*lh+[0..7] of column `col` in Wp
    const bf16_t* wzp = Wpz + (size_t)col * KDIM_ + 8 * lh;
    const bf16_t* wfp = Wpf + (size_t)col * KDIM_ + 8 * lh;
    // A (16x32 bf16): lane = row l; q[0] = K 8*lh+[0..7], q[1] = K 16+8*lh+[0..7]
    const char* abase = smem + l * RSTRIDE + lh * 16;

    v8f cz[4] = {v8f{}, v8f{}, v8f{}, v8f{}};
    v8f cf[4] = {v8f{}, v8f{}, v8f{}, v8f{}};

#pragma unroll
    for (int half = 0; half < 2; ++half) {                  // half 0: X[t-1] slices, 1: X[t] slices
        const bf16_t* wz = wzp + (half << 10);
        const bf16_t* wf = wfp + (half << 10);
        const char*   p  = abase + half * SSTRIDE;          // slice select folds into immediates
        for (int ko = 0; ko < 1024; ko += 32, p += GSTRIDE) {
            union { v16bf v; uint4 q[2]; } Bz, Bf, A;
            Bz.q[0] = *(const uint4*)(wz + ko);
            Bz.q[1] = *(const uint4*)(wz + ko + 16);
            Bf.q[0] = *(const uint4*)(wf + ko);
            Bf.q[1] = *(const uint4*)(wf + ko + 16);

            A.q[0] = *(const uint4*)(p);
            A.q[1] = *(const uint4*)(p + 32);
            cz[0] = __builtin_amdgcn_wmma_f32_16x16x32_bf16(false, A.v, false, Bz.v, (short)0, cz[0], false, false);
            cf[0] = __builtin_amdgcn_wmma_f32_16x16x32_bf16(false, A.v, false, Bf.v, (short)0, cf[0], false, false);

            A.q[0] = *(const uint4*)(p + 1 * SSTRIDE);
            A.q[1] = *(const uint4*)(p + 1 * SSTRIDE + 32);
            cz[1] = __builtin_amdgcn_wmma_f32_16x16x32_bf16(false, A.v, false, Bz.v, (short)0, cz[1], false, false);
            cf[1] = __builtin_amdgcn_wmma_f32_16x16x32_bf16(false, A.v, false, Bf.v, (short)0, cf[1], false, false);

            A.q[0] = *(const uint4*)(p + 2 * SSTRIDE);
            A.q[1] = *(const uint4*)(p + 2 * SSTRIDE + 32);
            cz[2] = __builtin_amdgcn_wmma_f32_16x16x32_bf16(false, A.v, false, Bz.v, (short)0, cz[2], false, false);
            cf[2] = __builtin_amdgcn_wmma_f32_16x16x32_bf16(false, A.v, false, Bf.v, (short)0, cf[2], false, false);

            A.q[0] = *(const uint4*)(p + 3 * SSTRIDE);
            A.q[1] = *(const uint4*)(p + 3 * SSTRIDE + 32);
            cz[3] = __builtin_amdgcn_wmma_f32_16x16x32_bf16(false, A.v, false, Bz.v, (short)0, cz[3], false, false);
            cf[3] = __builtin_amdgcn_wmma_f32_16x16x32_bf16(false, A.v, false, Bf.v, (short)0, cf[3], false, false);
        }
    }

    const float bzl = bz[col];
    const float bfl = bfv[col];
#pragma unroll
    for (int j = 0; j < 4; ++j) {
        const size_t rowbase = (size_t)(t0 + j) * BATCH_;
#pragma unroll
        for (int r = 0; r < 8; ++r) {
            const int m = r + 8 * lh;                       // D: VGPR r holds M=r (lanes 0-15), M=r+8 (16-31)
            float z = cz[j][r] + bzl;
            float f = cf[j][r] + bfl;
            z = z / (1.0f + __expf(-1.702f * z));           // quick_gelu
            f = 1.0f / (1.0f + __expf(-f));                 // sigmoid
            const size_t o = (rowbase + m) * HID_ + col;    // == t*16384 + b*1024 + h
            Zact[o] = (__bf16)z;
            Fact[o] = (__bf16)f;
        }
    }
}

// ---------------------------------------------------------------- scan pass A: per-chunk carries
// h_t = a_t*h_{t-1} + b_t with a = 1-f, b = f*z.  Chunk carry: Ac = prod a, Bc = chunk result from h=0.
__global__ __launch_bounds__(256) void qrnn_scan_carry(const bf16_t* __restrict__ Z,
                                                       const bf16_t* __restrict__ F,
                                                       float* __restrict__ Ac,
                                                       float* __restrict__ Bc) {
    const int    c   = blockIdx.y;                          // chunk 0..63
    const size_t idx = ((size_t)blockIdx.x * 256 + threadIdx.x) * 4;
    float a0 = 1.f, a1 = 1.f, a2 = 1.f, a3 = 1.f;
    float b0 = 0.f, b1 = 0.f, b2 = 0.f, b3 = 0.f;
    for (int t = c * LCH_; t < c * LCH_ + LCH_; ++t) {
        const size_t o = (size_t)t * LANES_ + idx;
        const v4bf zv = *(const v4bf*)(Z + o);
        const v4bf fv = *(const v4bf*)(F + o);
        const float f0 = (float)fv[0], f1 = (float)fv[1], f2 = (float)fv[2], f3 = (float)fv[3];
        b0 = fmaf(f0, (float)zv[0] - b0, b0);  a0 *= (1.f - f0);
        b1 = fmaf(f1, (float)zv[1] - b1, b1);  a1 *= (1.f - f1);
        b2 = fmaf(f2, (float)zv[2] - b2, b2);  a2 *= (1.f - f2);
        b3 = fmaf(f3, (float)zv[3] - b3, b3);  a3 *= (1.f - f3);
    }
    const size_t co = (size_t)c * LANES_ + idx;
    *(float4*)(Ac + co) = float4{a0, a1, a2, a3};
    *(float4*)(Bc + co) = float4{b0, b1, b2, b3};
}

// ---------------------------------------------------------------- scan pass B: combine chunk carries
__global__ __launch_bounds__(256) void qrnn_scan_combine(const float* __restrict__ Ac,
                                                         const float* __restrict__ Bc,
                                                         float* __restrict__ S) {
    const size_t idx = ((size_t)blockIdx.x * 256 + threadIdx.x) * 4;
    float s0 = 0.f, s1 = 0.f, s2 = 0.f, s3 = 0.f;
    for (int c = 0; c < NCHUNK_; ++c) {
        const size_t co = (size_t)c * LANES_ + idx;
        *(float4*)(S + co) = float4{s0, s1, s2, s3};        // state at chunk start
        const float4 a = *(const float4*)(Ac + co);
        const float4 b = *(const float4*)(Bc + co);
        s0 = fmaf(a.x, s0, b.x);
        s1 = fmaf(a.y, s1, b.y);
        s2 = fmaf(a.z, s2, b.z);
        s3 = fmaf(a.w, s3, b.w);
    }
}

// ---------------------------------------------------------------- scan pass C: expand chunks, write H
__global__ __launch_bounds__(256) void qrnn_scan_expand(const bf16_t* __restrict__ Z,
                                                        const bf16_t* __restrict__ F,
                                                        const float* __restrict__ S,
                                                        float* __restrict__ out) {
    const int    c   = blockIdx.y;
    const size_t idx = ((size_t)blockIdx.x * 256 + threadIdx.x) * 4;
    const float4 s = *(const float4*)(S + (size_t)c * LANES_ + idx);
    float h0 = s.x, h1 = s.y, h2 = s.z, h3 = s.w;
    for (int t = c * LCH_; t < c * LCH_ + LCH_; ++t) {
        const size_t o = (size_t)t * LANES_ + idx;
        const v4bf zv = *(const v4bf*)(Z + o);
        const v4bf fv = *(const v4bf*)(F + o);
        const float f0 = (float)fv[0], f1 = (float)fv[1], f2 = (float)fv[2], f3 = (float)fv[3];
        h0 = fmaf(f0, (float)zv[0] - h0, h0);               // f*z + (1-f)*h
        h1 = fmaf(f1, (float)zv[1] - h1, h1);
        h2 = fmaf(f2, (float)zv[2] - h2, h2);
        h3 = fmaf(f3, (float)zv[3] - h3, h3);
        *(float4*)(out + o) = float4{h0, h1, h2, h3};
    }
    if (c == NCHUNK_ - 1)                                   // H[-1] tail
        *(float4*)(out + (size_t)SEQ_ * LANES_ + idx) = float4{h0, h1, h2, h3};
}

// ---------------------------------------------------------------- launcher
extern "C" void kernel_launch(void* const* d_in, const int* in_sizes, int n_in,
                              void* d_out, int out_size, void* d_ws, size_t ws_size,
                              hipStream_t stream) {
    const float* X   = (const float*)d_in[0];
    const float* Wz  = (const float*)d_in[1];
    const float* bz  = (const float*)d_in[2];
    const float* Wf  = (const float*)d_in[3];
    const float* bfp = (const float*)d_in[4];

    char*   ws  = (char*)d_ws;
    bf16_t* Xb  = (bf16_t*)(ws + XB_OFF);
    bf16_t* Wpz = (bf16_t*)(ws + WPZ_OFF);
    bf16_t* Wpf = (bf16_t*)(ws + WPF_OFF);
    bf16_t* Za  = (bf16_t*)(ws + ZA_OFF);
    bf16_t* Fa  = (bf16_t*)(ws + FA_OFF);
    float*  Ac  = (float*)(ws + AC_OFF);
    float*  Bc  = (float*)(ws + BC_OFF);
    float*  S   = (float*)(ws + SS_OFF);

    // 1) X -> bf16 (16.7M elems, 4/thread)
    qrnn_pack_x<<<16384, 256, 0, stream>>>(X, Xb);
    // 2) weights -> bf16 packed (H, kk) for both z and f
    qrnn_pack_w<<<dim3(8192, 2), 256, 0, stream>>>(Wz, Wf, Wpz, Wpf);
    // 3) fused dual-GEMM + bias + activations (202 KB dynamic LDS per block)
    qrnn_gemm<<<dim3(8, 256), 256, LDS_BYTES, stream>>>(Xb, Wpz, Wpf, bz, bfp, Za, Fa);
    // 4) parallel linear-recurrence scan: carries -> combine -> expand
    qrnn_scan_carry<<<dim3(16, NCHUNK_), 256, 0, stream>>>(Za, Fa, Ac, Bc);
    qrnn_scan_combine<<<16, 256, 0, stream>>>(Ac, Bc, S);
    qrnn_scan_expand<<<dim3(16, NCHUNK_), 256, 0, stream>>>(Za, Fa, S, (float*)d_out);
}
